// InstantNGP_76132590289312
// MI455X (gfx1250) — matrix operand order repaired and checked by reading, not compiled, not run on hardware
//
#include <hip/hip_runtime.h>

typedef __attribute__((ext_vector_type(16))) _Float16 v16h;
typedef __attribute__((ext_vector_type(8)))  _Float16 v8h;
typedef __attribute__((ext_vector_type(8)))  float    v8f;

#define NUM_LEVELS 16

struct NGPParams {
  const float* pos;
  const float* tbl[NUM_LEVELS];
  const float* w1; const float* b1;
  const float* w2; const float* b2;
  const float* w3; const float* b3;
  float* out;          // rgb at [0, 3N), density at [3N, 4N)
  int n_points;
};

__device__ __forceinline__ v8f wmma16x16x32(v16h a, v16h b, v8f c) {
  return __builtin_amdgcn_wmma_f32_16x16x32_f16(false, a, false, b, (short)0, c,
                                                false, false);
}

// single-instruction ReLU: v_med3_f32(x, 0, +inf) == clamp(x, 0, inf)
__device__ __forceinline__ float relu1(float x) {
  return __builtin_amdgcn_fmed3f(x, 0.0f, __builtin_inff());
}

// Read one 32-row K-slice of a B fragment column from LDS: two runs of 8
// halfs (k0..k0+7, k0+16..k0+23) at 16B alignment -> two ds_load_b128.
__device__ __forceinline__ v16h load_bfrag(const _Float16* colbase, int k0) {
  v8h lo = *(const v8h*)(colbase + k0);
  v8h hi = *(const v8h*)(colbase + k0 + 16);
  return __builtin_shufflevector(lo, hi, 0, 1, 2, 3, 4, 5, 6, 7,
                                 8, 9, 10, 11, 12, 13, 14, 15);
}

__launch_bounds__(256)
__global__ void ngp_fused_wmma_kernel(NGPParams P) {
  // level resolutions r = int(16 * 128**(l/15)); table sizes = min(r^3, 2^19)
  constexpr uint32_t kRes[16] = {16, 22, 30, 42, 58, 80, 111, 153,
                                 212, 294, 406, 561, 776, 1072, 1482, 2048};
  constexpr uint32_t kTsz[16] = {4096, 10648, 27000, 74088, 195112, 512000,
                                 524288, 524288, 524288, 524288, 524288,
                                 524288, 524288, 524288, 524288, 524288};
  const uint32_t HC1 = 2654435761u, HC2 = 805459861u, MASK = (1u << 19) - 1u;

  // f16 weight copies, column-major [col][k], padded strides for bank spread
  __shared__ alignas(16) _Float16 wl1[64 * 40];       // K=32, stride 40
  __shared__ alignas(16) _Float16 wl2[64 * 72];       // K=64, stride 72
  __shared__ alignas(16) _Float16 wl3[16 * 72];       // K=64 (cols 4..15 zero)
  // per-wave staging buffer for inter-layer transposes
  __shared__ alignas(16) _Float16 hbuf[8][16][72];

  const int lane = threadIdx.x & 31;
  const int wv   = threadIdx.x >> 5;
  const int row  = lane & 15;   // point-in-tile / matrix row-or-col id
  const int half = lane >> 4;   // 0: lanes 0-15, 1: lanes 16-31

  // ---- cooperative f32 -> f16 weight staging (coalesced reads) ----------
  for (int idx = threadIdx.x; idx < 64 * 32; idx += 256) {
    const int k = idx >> 6, col = idx & 63;          // idx == k*64 + col
    wl1[col * 40 + k] = (_Float16)P.w1[idx];
  }
  for (int idx = threadIdx.x; idx < 64 * 64; idx += 256) {
    const int k = idx >> 6, col = idx & 63;
    wl2[col * 72 + k] = (_Float16)P.w2[idx];
  }
  for (int idx = threadIdx.x; idx < 16 * 72; idx += 256) {
    const int col = idx / 72, k = idx % 72;
    const float v = (col < 4 && k < 64) ? P.w3[k * 4 + col] : 0.0f;
    wl3[idx] = (_Float16)v;
  }
  __syncthreads();

  float b1v[4], b2v[4];
#pragma unroll
  for (int t = 0; t < 4; ++t) {
    b1v[t] = P.b1[t * 16 + row];
    b2v[t] = P.b2[t * 16 + row];
  }
  const float b3v = (row < 4) ? P.b3[row] : 0.0f;

  const int ntiles   = P.n_points >> 4;
  const int gw       = blockIdx.x * 8 + wv;
  const int nwaves   = gridDim.x * 8;
  const int dens_off = 3 * P.n_points;

  // per-lane B-fragment column bases (uniform over the tile loop)
  const _Float16* w1c[4];
  const _Float16* w2c[4];
#pragma unroll
  for (int t = 0; t < 4; ++t) {
    w1c[t] = &wl1[(t * 16 + row) * 40];
    w2c[t] = &wl2[(t * 16 + row) * 72];
  }
  const _Float16* w3c = &wl3[row * 72];

  // -------- gather stage: hash + 8 float2 table loads for one tile -------
  // A layout (16-bit 16x32): lane<16: row=lane, halfs0-7=K0..7, 8-15=K16..23
  //                          lane>=16: row=lane-16, K8..15 and K24..31
  // K = 2*level + f: half 0 gathers levels 0-3 & 8-11, half 1 -> 4-7 & 12-15
  auto gather = [&](int tile, float2 fv[8]) {
    const int p = (tile << 4) + row;
    const float px = P.pos[3 * p + 0];
    const float py = P.pos[3 * p + 1];
    const float pz = P.pos[3 * p + 2];
#pragma unroll
    for (int g = 0; g < 2; ++g) {
#pragma unroll
      for (int j = 0; j < 4; ++j) {
        const int lvl = g * 8 + half * 4 + j;
        const uint32_t r = kRes[lvl];
        const uint32_t cx = (uint32_t)(px * (float)r);
        const uint32_t cy = (uint32_t)(py * (float)r);
        const uint32_t cz = (uint32_t)(pz * (float)r);
        const uint32_t h = (cx ^ (cy * HC1) ^ (cz * HC2)) & MASK;
        const uint32_t idx = h % kTsz[lvl];   // folds to AND / magic-mul
        fv[g * 4 + j] = ((const float2*)P.tbl[lvl])[idx];
      }
    }
  };

  // software pipeline: gathers for tile t+stride are in flight while the
  // MLP of tile t runs (loads retire in order -> partial loadcnt waits).
  float2 fv[8];
  if (gw < ntiles) gather(gw, fv);

  for (int tile = gw; tile < ntiles; tile += nwaves) {
    const int next = tile + nwaves;
    float2 fnext[8];
    if (next < ntiles) gather(next, fnext);   // issue next-tile gathers early

    // pack this tile's features into the A fragment
    v16h A;
#pragma unroll
    for (int i = 0; i < 8; ++i) {
      A[i * 2 + 0] = (_Float16)fv[i].x;
      A[i * 2 + 1] = (_Float16)fv[i].y;
    }

    // -------- layer 1: [16x32] x [32x64] -> 4 WMMAs ----------------------
    v8f acc1[4];
#pragma unroll
    for (int t = 0; t < 4; ++t) {
      v8f c;
#pragma unroll
      for (int m = 0; m < 8; ++m) c[m] = b1v[t];
      acc1[t] = wmma16x16x32(A, load_bfrag(w1c[t], half * 8), c);
    }
    // ReLU + transpose C-layout -> row-major f16 in LDS
#pragma unroll
    for (int t = 0; t < 4; ++t)
#pragma unroll
      for (int m = 0; m < 8; ++m)
        hbuf[wv][half * 8 + m][t * 16 + row] = (_Float16)relu1(acc1[t][m]);

    // re-read as two 16x32 A fragments (contiguous 16B ds_load_b128 pairs)
    v16h A2[2];
#pragma unroll
    for (int c = 0; c < 2; ++c) {
      v8h lo = *(const v8h*)&hbuf[wv][row][c * 32 + half * 8];
      v8h hi = *(const v8h*)&hbuf[wv][row][c * 32 + 16 + half * 8];
      A2[c] = __builtin_shufflevector(lo, hi, 0, 1, 2, 3, 4, 5, 6, 7,
                                      8, 9, 10, 11, 12, 13, 14, 15);
    }

    // -------- layer 2: [16x64] x [64x64] -> 8 WMMAs ----------------------
    v8f acc2[4];
#pragma unroll
    for (int t = 0; t < 4; ++t) {
      v8f c;
#pragma unroll
      for (int m = 0; m < 8; ++m) c[m] = b2v[t];
      c = wmma16x16x32(A2[0], load_bfrag(w2c[t], half * 8), c);
      acc2[t] = wmma16x16x32(A2[1], load_bfrag(w2c[t], 32 + half * 8), c);
    }
#pragma unroll
    for (int t = 0; t < 4; ++t)
#pragma unroll
      for (int m = 0; m < 8; ++m)
        hbuf[wv][half * 8 + m][t * 16 + row] = (_Float16)relu1(acc2[t][m]);

    v16h A3[2];
#pragma unroll
    for (int c = 0; c < 2; ++c) {
      v8h lo = *(const v8h*)&hbuf[wv][row][c * 32 + half * 8];
      v8h hi = *(const v8h*)&hbuf[wv][row][c * 32 + 16 + half * 8];
      A3[c] = __builtin_shufflevector(lo, hi, 0, 1, 2, 3, 4, 5, 6, 7,
                                      8, 9, 10, 11, 12, 13, 14, 15);
    }

    // -------- layer 3: [16x64] x [64x16(pad)] -> 2 WMMAs -----------------
    v8f acc3;
    {
      v8f c;
#pragma unroll
      for (int m = 0; m < 8; ++m) c[m] = b3v;
      c = wmma16x16x32(A3[0], load_bfrag(w3c, half * 8), c);
      acc3 = wmma16x16x32(A3[1], load_bfrag(w3c, 32 + half * 8), c);
    }

    // -------- epilogue: cols 0-2 -> sigmoid rgb, col 3 -> softplus density
    // D layout: lane holds col=row, VGPR m = matrix row (half*8+m)
    const int pbase = tile << 4;
    if (row < 3) {
#pragma unroll
      for (int m = 0; m < 8; ++m) {
        const float v = acc3[m];
        P.out[(pbase + half * 8 + m) * 3 + row] =
            __builtin_amdgcn_rcpf(1.0f + __expf(-v));
      }
    } else if (row == 3) {
#pragma unroll
      for (int m = 0; m < 8; ++m) {
        const float v = acc3[m];
        P.out[dens_off + pbase + half * 8 + m] =
            relu1(v) + log1pf(__expf(-fabsf(v)));
      }
    }

    // rotate the pipeline
#pragma unroll
    for (int i = 0; i < 8; ++i) fv[i] = fnext[i];
  }
}

extern "C" void kernel_launch(void* const* d_in, const int* in_sizes, int n_in,
                              void* d_out, int out_size, void* d_ws,
                              size_t ws_size, hipStream_t stream) {
  (void)n_in; (void)out_size; (void)d_ws; (void)ws_size;
  NGPParams P;
  P.pos = (const float*)d_in[0];
  for (int l = 0; l < NUM_LEVELS; ++l) P.tbl[l] = (const float*)d_in[1 + l];
  P.w1 = (const float*)d_in[17];
  P.b1 = (const float*)d_in[18];
  P.w2 = (const float*)d_in[19];
  P.b2 = (const float*)d_in[20];
  P.w3 = (const float*)d_in[21];
  P.b3 = (const float*)d_in[22];
  P.out = (float*)d_out;
  P.n_points = in_sizes[0] / 3;

  // 2048 blocks x 8 waves = 16384 waves; 131072 tiles -> 8 tiles/wave.
  ngp_fused_wmma_kernel<<<dim3(2048), dim3(256), 0, stream>>>(P);
}
